// CDCNet_78864189489969
// MI455X (gfx1250) — matrix operand order repaired and checked by reading, not compiled, not run on hardware
//
#include <hip/hip_runtime.h>

typedef __attribute__((ext_vector_type(16))) _Float16 v16h;
typedef __attribute__((ext_vector_type(8)))  _Float16 v8h;
typedef __attribute__((ext_vector_type(8)))  float    v8f;

__device__ __forceinline__ float eluf(float x)      { return x > 0.f ? x : (__expf(x) - 1.f); }
__device__ __forceinline__ float sigmoidf_(float x) { return 1.f / (1.f + __expf(-x)); }

// ---------------------------------------------------------------------------
// WMMA GEMM:  Y[nrows, DOUT] (+)= act( X[nrows, DIN] @ W[DIN, DOUT] + bias )
// One wave per 16x16 output tile; nrows assumed multiple of 16 (N=10000).
// Compile-time DIN/DOUT: K-loop fully unrolled, guards constant-folded.
//   A fragment: direct global float4 loads (two 8-consecutive-K runs per lane)
//   B fragment: staged to LDS column-major, consumed as 2x ds_load_b128
// ---------------------------------------------------------------------------
template<int DIN, int DOUT, int ACT, int ACCUM>
__global__ void gemm_wmma(const float* __restrict__ X, int ldx,
                          const float* __restrict__ W,
                          const float* __restrict__ bias,
                          float* __restrict__ Y, int ldy)
{
    // B tile, column-major: Bs[n*32 + k], n in [0,16), k in [0,32)
    __shared__ __align__(16) _Float16 Bs[16 * 32];

    const int L   = threadIdx.x;
    const int r16 = L & 15;
    const int g   = L >> 4;
    const int m0  = blockIdx.x * 16;
    const int n0  = blockIdx.y * 16;
    const int n   = n0 + r16;            // this lane's output column (all 8 C regs)
    constexpr bool FULL_N = (DOUT % 16 == 0);

    const float* __restrict__ xrow = X + (size_t)(m0 + r16) * ldx;
    float* __restrict__ yp = Y + (size_t)(m0 + 8 * g) * ldy + n;  // rows r=0..7 at yp + r*ldy

    v8f c = {};
    if (ACCUM) {
        if (FULL_N || n < DOUT) {
#pragma unroll
            for (int r = 0; r < 8; ++r) c[r] = yp[(size_t)r * ldy];
        }
    }

#pragma unroll
    for (int ks = 0; ks < DIN; ks += 32) {
        // ---- A fragment: lane needs K = ks+g*8+{0..7} and ks+16+g*8+{0..7}
        const bool hiValid = (ks + 32 <= DIN);          // compile-time per step
        float4 a0 = *(const float4*)(xrow + ks + g * 8);
        float4 a1 = *(const float4*)(xrow + ks + g * 8 + 4);
        float4 a2 = make_float4(0.f, 0.f, 0.f, 0.f);
        float4 a3 = make_float4(0.f, 0.f, 0.f, 0.f);
        if (hiValid) {
            a2 = *(const float4*)(xrow + ks + 16 + g * 8);
            a3 = *(const float4*)(xrow + ks + 16 + g * 8 + 4);
        }
        v16h a;
        a[0]  = (_Float16)a0.x; a[1]  = (_Float16)a0.y;
        a[2]  = (_Float16)a0.z; a[3]  = (_Float16)a0.w;
        a[4]  = (_Float16)a1.x; a[5]  = (_Float16)a1.y;
        a[6]  = (_Float16)a1.z; a[7]  = (_Float16)a1.w;
        a[8]  = (_Float16)a2.x; a[9]  = (_Float16)a2.y;
        a[10] = (_Float16)a2.z; a[11] = (_Float16)a2.w;
        a[12] = (_Float16)a3.x; a[13] = (_Float16)a3.y;
        a[14] = (_Float16)a3.z; a[15] = (_Float16)a3.w;

        // ---- stage B: lane L loads W row (ks+L), cols n0..n0+15, scatters
        //      into column-major LDS tile
        {
            const bool rowValid = (ks + 32 <= DIN) || (ks + L < DIN);
            float wv[16];
            if (rowValid) {
                const float* __restrict__ wrow = W + (size_t)(ks + L) * DOUT + n0;
#pragma unroll
                for (int t = 0; t < 16; t += 4) {
                    float4 w4 = make_float4(0.f, 0.f, 0.f, 0.f);
                    if (n0 + t + 3 < DOUT) w4 = *(const float4*)(wrow + t);
                    wv[t + 0] = w4.x; wv[t + 1] = w4.y;
                    wv[t + 2] = w4.z; wv[t + 3] = w4.w;
                }
            } else {
#pragma unroll
                for (int t = 0; t < 16; ++t) wv[t] = 0.f;
            }
#pragma unroll
            for (int t = 0; t < 16; ++t)
                Bs[t * 32 + L] = (_Float16)wv[t];
        }
        __syncthreads();

        // ---- B fragment: column r16, K = g*8+{0..7} and 16+g*8+{0..7}
        const v8h* bp = reinterpret_cast<const v8h*>(&Bs[r16 * 32 + g * 8]);
        v8h b0 = bp[0], b1 = bp[2];
        v16h b = __builtin_shufflevector(b0, b1,
                                         0, 1, 2, 3, 4, 5, 6, 7,
                                         8, 9, 10, 11, 12, 13, 14, 15);

        c = __builtin_amdgcn_wmma_f32_16x16x32_f16(false, a, false, b,
                                                   (short)0, c, false, false);
        __syncthreads();
    }

    if (FULL_N || n < DOUT) {
        float bv = (bias != nullptr) ? bias[n] : 0.f;
#pragma unroll
        for (int r = 0; r < 8; ++r) {
            float v = c[r] + bv;
            if (ACT == 1) v = eluf(v);
            yp[(size_t)r * ldy] = v;
        }
    }
}

template<int DIN, int DOUT, int ACT, int ACCUM>
static void launch_gemm(const float* X, int ldx, const float* W,
                        const float* bias, float* Y, int ldy,
                        int nrows, hipStream_t stream)
{
    dim3 grid(nrows / 16, (DOUT + 15) / 16), block(32);
    gemm_wmma<DIN, DOUT, ACT, ACCUM><<<grid, block, 0, stream>>>(
        X, ldx, W, bias, Y, ldy);
}

// ---------------------------------------------------------------------------
// BatchNorm1d (training mode: batch mean, biased variance), one block per dim
// ---------------------------------------------------------------------------
__global__ void bn_kernel(const float* __restrict__ x, int ldx,
                          float* __restrict__ y, int ldy,
                          const float* __restrict__ gw,
                          const float* __restrict__ bw, int n)
{
    const int d = blockIdx.x;
    const int tid = threadIdx.x;
    __shared__ float r1[256], r2[256];
    float s = 0.f, s2 = 0.f;
    for (int i = tid; i < n; i += 256) {
        float v = x[(size_t)i * ldx + d];
        s += v; s2 += v * v;
    }
    r1[tid] = s; r2[tid] = s2;
    __syncthreads();
    for (int off = 128; off > 0; off >>= 1) {
        if (tid < off) { r1[tid] += r1[tid + off]; r2[tid] += r2[tid + off]; }
        __syncthreads();
    }
    float mean = r1[0] / (float)n;
    float var  = r2[0] / (float)n - mean * mean;
    float inv  = rsqrtf(var + 1e-5f);
    float gg = gw[d] * inv;
    float bb = bw[d] - mean * gg;
    for (int i = tid; i < n; i += 256)
        y[(size_t)i * ldy + d] = x[(size_t)i * ldx + d] * gg + bb;
}

// Column mean over N rows, one block per dim
__global__ void colmean_kernel(const float* __restrict__ x, int ldx,
                               float* __restrict__ mean, int n)
{
    const int d = blockIdx.x;
    const int tid = threadIdx.x;
    __shared__ float r1[256];
    float s = 0.f;
    for (int i = tid; i < n; i += 256) s += x[(size_t)i * ldx + d];
    r1[tid] = s;
    __syncthreads();
    for (int off = 128; off > 0; off >>= 1) {
        if (tid < off) r1[tid] += r1[tid + off];
        __syncthreads();
    }
    if (tid == 0) mean[d] = r1[0] / (float)n;
}

// Effective l0 bias: cb[j] = b0[j] + sum_d mean[d] * W0[(D+d)*64 + j]
// (global-exchange concat folded into a constant bias)
__global__ void l0bias_kernel(const float* __restrict__ mean,
                              const float* __restrict__ W0,
                              const float* __restrict__ b0,
                              float* __restrict__ cb, int d)
{
    int j = threadIdx.x;   // 64 threads
    float acc = b0[j];
    for (int t = 0; t < d; ++t) acc += mean[t] * W0[(size_t)(d + t) * 64 + j];
    cb[j] = acc;
}

// ---------------------------------------------------------------------------
// GravNet: per-node top-10 nearest neighbors in 4-D latent space (streamed
// through LDS as float4), weighted message mean/max aggregation.
// agg[i, 0:64] = mean_k(h[idx]*w),  agg[i, 64:128] = max_k(h[idx]*w)
// ---------------------------------------------------------------------------
#define GN_CHUNK 1024
__global__ void gravnet_kernel(const float* __restrict__ s,
                               const float* __restrict__ h,
                               float* __restrict__ agg, int n)
{
    __shared__ __align__(16) float ss[GN_CHUNK * 4];
    const int i = blockIdx.x * blockDim.x + threadIdx.x;
    float4 si = make_float4(0.f, 0.f, 0.f, 0.f);
    if (i < n) si = *(const float4*)(s + (size_t)i * 4);

    float bd[10]; int bi[10];
#pragma unroll
    for (int t = 0; t < 10; ++t) { bd[t] = 3.4e38f; bi[t] = 0; }

    for (int base = 0; base < n; base += GN_CHUNK) {
        int cnt = min(GN_CHUNK, n - base);
        for (int t = threadIdx.x; t < cnt; t += blockDim.x)
            ((float4*)ss)[t] = ((const float4*)(s + (size_t)base * 4))[t];
        __syncthreads();
        if (i < n) {
            for (int j = 0; j < cnt; ++j) {
                float4 sj = *(const float4*)(ss + j * 4);
                float d0 = sj.x - si.x, d1 = sj.y - si.y;
                float d2 = sj.z - si.z, d3 = sj.w - si.w;
                float dv = d0*d0 + d1*d1 + d2*d2 + d3*d3;
                if (dv < bd[9]) {
                    int iv = base + j;
#pragma unroll
                    for (int p = 0; p < 10; ++p) {      // bubble-insert, keeps sorted
                        if (dv < bd[p]) {
                            float td = bd[p]; int ti = bi[p];
                            bd[p] = dv; bi[p] = iv; dv = td; iv = ti;
                        }
                    }
                }
            }
        }
        __syncthreads();
    }
    if (i >= n) return;

    float w[10];
#pragma unroll
    for (int t = 0; t < 10; ++t) w[t] = __expf(-10.f * bd[t]);

    for (int d = 0; d < 64; ++d) {
        float sm = 0.f, mx = -3.4e38f;
#pragma unroll
        for (int t = 0; t < 10; ++t) {
            float m = h[(size_t)bi[t] * 64 + d] * w[t];
            sm += m; mx = fmaxf(mx, m);
        }
        agg[(size_t)i * 128 + d]      = sm * 0.1f;   // mean over k=10
        agg[(size_t)i * 128 + 64 + d] = mx;
    }
}

// ---------------------------------------------------------------------------
// Output heads: beta(sigmoid), coords, p3, vertex, charge(sigmoid)
// ---------------------------------------------------------------------------
__global__ void heads_kernel(const float* __restrict__ z,
                             const float* Wb, const float* bb,
                             const float* Wc, const float* bc,
                             const float* Wp, const float* bp,
                             const float* Wv, const float* bv,
                             const float* Wq, const float* bq,
                             float* __restrict__ out, int n)
{
    int i = blockIdx.x * blockDim.x + threadIdx.x;
    if (i >= n) return;
    float zz[64];
#pragma unroll
    for (int d = 0; d < 64; ++d) zz[d] = z[(size_t)i * 64 + d];

    float accb = bb[0], accq = bq[0];
    float ac0 = bc[0], ac1 = bc[1];
    float ap0 = bp[0], ap1 = bp[1], ap2 = bp[2];
    float av0 = bv[0], av1 = bv[1], av2 = bv[2];
#pragma unroll
    for (int d = 0; d < 64; ++d) {
        float zd = zz[d];
        accb += zd * Wb[d];
        accq += zd * Wq[d];
        ac0 += zd * Wc[d*2+0]; ac1 += zd * Wc[d*2+1];
        ap0 += zd * Wp[d*3+0]; ap1 += zd * Wp[d*3+1]; ap2 += zd * Wp[d*3+2];
        av0 += zd * Wv[d*3+0]; av1 += zd * Wv[d*3+1]; av2 += zd * Wv[d*3+2];
    }
    out[i] = sigmoidf_(accb);
    out[n + i*2 + 0] = ac0;  out[n + i*2 + 1] = ac1;
    out[3*n + i*3 + 0] = ap0; out[3*n + i*3 + 1] = ap1; out[3*n + i*3 + 2] = ap2;
    out[6*n + i*3 + 0] = av0; out[6*n + i*3 + 1] = av1; out[6*n + i*3 + 2] = av2;
    out[9*n + i] = sigmoidf_(accq);
}

// ---------------------------------------------------------------------------
extern "C" void kernel_launch(void* const* d_in, const int* in_sizes, int n_in,
                              void* d_out, int out_size, void* d_ws, size_t ws_size,
                              hipStream_t stream)
{
    (void)n_in; (void)out_size; (void)ws_size;
    const int INPUT_DIM = 16, NB = 4;
    const int N = in_sizes[0] / INPUT_DIM;   // 10000, multiple of 16
    const float* x_in = (const float*)d_in[0];
    auto F = [&](int i) { return (const float*)d_in[i]; };

    float* ws = (float*)d_ws;
    size_t off = 0;
    auto alloc = [&](size_t cnt) { float* p = ws + off; off += cnt; return p; };
    float* xcur  = alloc((size_t)N * 128);
    float* t0    = alloc((size_t)N * 64);
    float* t1    = alloc((size_t)N * 64);
    float* sbuf  = alloc((size_t)N * 4);
    float* hbuf  = alloc((size_t)N * 64);
    float* aggb  = alloc((size_t)N * 128);
    float* x2    = alloc((size_t)N * 128);
    float* feats = alloc((size_t)N * 128);
    float* zbuf  = alloc((size_t)N * 64);
    float* meanb = alloc(128);
    float* cbias = alloc(64);

    // bn0 on raw input -> xcur (ld=16)
    bn_kernel<<<dim3(INPUT_DIM), dim3(256), 0, stream>>>(
        x_in, INPUT_DIM, xcur, INPUT_DIM, F(1), F(2), N);

    int dcur = INPUT_DIM;
    for (int blk = 0; blk < NB; ++blk) {
        int base = 3 + 19 * blk;
        // global-exchange mean -> effective l0 bias
        colmean_kernel<<<dim3(dcur), dim3(256), 0, stream>>>(xcur, dcur, meanb, N);
        l0bias_kernel<<<dim3(1), dim3(64), 0, stream>>>(
            meanb, F(base + 0), F(base + 1), cbias, dcur);
        if (blk == 0)
            launch_gemm<16, 64, 1, 0>(xcur, dcur, F(base + 0), cbias, t0, 64, N, stream);   // l0+elu
        else
            launch_gemm<128, 64, 1, 0>(xcur, dcur, F(base + 0), cbias, t0, 64, N, stream);  // l0+elu
        launch_gemm<64, 64, 1, 0>(t0, 64, F(base + 2), F(base + 3), t1, 64, N, stream);     // l1+elu
        bn_kernel<<<dim3(64), dim3(256), 0, stream>>>(
            t1, 64, t0, 64, F(base + 4), F(base + 5), N);                                    // bn2
        launch_gemm<64, 64, 1, 0>(t0, 64, F(base + 6), F(base + 7), t1, 64, N, stream);     // l3+elu (xg=t1)
        launch_gemm<64, 4, 0, 0>(t1, 64, F(base + 8), F(base + 9), sbuf, 4, N, stream);     // s latent
        launch_gemm<64, 64, 0, 0>(t1, 64, F(base + 10), F(base + 11), hbuf, 64, N, stream); // h msgs
        gravnet_kernel<<<dim3((N + 255) / 256), dim3(256), 0, stream>>>(sbuf, hbuf, aggb, N);
        launch_gemm<128, 128, 0, 0>(aggb, 128, F(base + 13), F(base + 14), x2, 128, N, stream); // o2(agg)
        launch_gemm<64, 128, 0, 1>(t1, 64, F(base + 12), nullptr, x2, 128, N, stream);          // += o1(x)
        bn_kernel<<<dim3(128), dim3(256), 0, stream>>>(
            x2, 128, xcur, 128, F(base + 15), F(base + 16), N);                                  // bn5
        launch_gemm<128, 32, 0, 0>(xcur, 128, F(base + 17), F(base + 18),
                                   feats + 32 * blk, 128, N, stream);                            // l6 -> feats col
        dcur = 128;
    }

    launch_gemm<128, 64, 1, 0>(feats, 128, F(79), F(80), zbuf, 64, N, stream);  // dense_cat+elu
    heads_kernel<<<dim3((N + 255) / 256), dim3(256), 0, stream>>>(
        zbuf, F(81), F(82), F(83), F(84), F(85), F(86),
        F(87), F(88), F(89), F(90), (float*)d_out, N);
}
